// Informer_55929064129361
// MI455X (gfx1250) — matrix operand order repaired and checked by reading, not compiled, not run on hardware
//
#include <hip/hip_runtime.h>
#include <math.h>

#define JAX_PARTITIONABLE 1

typedef _Float16 v16h  __attribute__((ext_vector_type(16)));
typedef _Float16 half8 __attribute__((ext_vector_type(8)));
typedef float    v8f   __attribute__((ext_vector_type(8)));

constexpr int B  = 8;
constexpr int DM = 512;
constexpr int NH = 8;
constexpr int DH = 64;
constexpr int DFF = 2048;

// ---------------------------------------------------------------- utilities

__device__ __forceinline__ float sinus_pe(int p, int c) {
    int j = c >> 1;
    float div = expf(-(float)(2 * j) * (9.210340371976184f / 512.0f)); // ln(10000)/512
    float ang = (float)p * div;
    return (c & 1) ? cosf(ang) : sinf(ang);
}

__device__ __forceinline__ void tf2x32(unsigned k0, unsigned k1, unsigned x0, unsigned x1,
                                       unsigned& o0, unsigned& o1) {
    unsigned ks2 = k0 ^ k1 ^ 0x1BD11BDAu;
    x0 += k0; x1 += k1;
#define TFR(d) { x0 += x1; x1 = (x1 << (d)) | (x1 >> (32 - (d))); x1 ^= x0; }
    TFR(13) TFR(15) TFR(26) TFR(6)   x0 += k1;  x1 += ks2 + 1u;
    TFR(17) TFR(29) TFR(16) TFR(24)  x0 += ks2; x1 += k0 + 2u;
    TFR(13) TFR(15) TFR(26) TFR(6)   x0 += k0;  x1 += k1 + 3u;
    TFR(17) TFR(29) TFR(16) TFR(24)  x0 += k1;  x1 += ks2 + 4u;
    TFR(13) TFR(15) TFR(26) TFR(6)   x0 += ks2; x1 += k0 + 5u;
#undef TFR
    o0 = x0; o1 = x1;
}

__device__ __forceinline__ float act_apply(float vv, int act) {
    if (act == 1)      vv = vv > 0.f ? vv : 0.f;
    else if (act == 2) vv = 0.5f * vv * (1.f + erff(vv * 0.70710678118654752f));
    return vv;
}

// ---------------------------------------------------------------- kernels

// data embedding: circular conv (7ch, 3 taps) + positional + temporal sinusoids
__global__ void k_embed(const float* __restrict__ x, const int* __restrict__ mark,
                        const float* __restrict__ cw, float* __restrict__ out, int L) {
    long i = (long)blockIdx.x * 256 + threadIdx.x;
    long total = (long)B * L * DM;
    if (i >= total) return;
    int o = (int)(i % DM);
    int t = (int)((i / DM) % L);
    int b = (int)(i / ((long)DM * L));
    float v = 0.f;
    for (int tap = 0; tap < 3; ++tap) {
        int tt = t - 1 + tap; if (tt < 0) tt += L; if (tt >= L) tt -= L;
        const float* xr = x + ((size_t)b * L + tt) * 7;
        for (int c = 0; c < 7; ++c) v += xr[c] * cw[(o * 7 + c) * 3 + tap];
    }
    const int* mk = mark + ((size_t)b * L + t) * 4;
    float pe = sinus_pe(t, o) + sinus_pe(mk[0], o) + sinus_pe(mk[1], o) +
               sinus_pe(mk[2], o) + sinus_pe(mk[3], o);
    out[i] = v + pe;
}

__global__ void k_f2h(const float* __restrict__ s, _Float16* __restrict__ d, long n) {
    long i = (long)blockIdx.x * 256 + threadIdx.x;
    if (i < n) d[i] = (_Float16)s[i];
}

__global__ void k_w2h(const float* __restrict__ W, _Float16* __restrict__ d,
                      int dout, int K, long n) {
    long i = (long)blockIdx.x * 256 + threadIdx.x;
    if (i >= n) return;
    int o = (int)(i / K);
    d[i] = (o < dout) ? (_Float16)W[i] : (_Float16)0.0f;
}

// pack conv weight (512,512,3) -> (512, 1536) with [tap0 | tap1 | tap2]
__global__ void k_packW(const float* __restrict__ W, _Float16* __restrict__ dst) {
    int i = blockIdx.x * 256 + threadIdx.x;
    if (i >= 512 * 1536) return;
    int o = i / 1536, j = i % 1536;
    int tap = j / 512, c = j % 512;
    dst[i] = (_Float16)W[(o * 512 + c) * 3 + tap];
}

// build circular 3-tap concat input (N, 1536) in f16
__global__ void k_build3(const float* __restrict__ X, _Float16* __restrict__ dst, int L) {
    long i = (long)blockIdx.x * 256 + threadIdx.x;
    long total = (long)B * L * 1536;
    if (i >= total) return;
    int j = (int)(i % 1536);
    long row = i / 1536;
    int t = (int)(row % L), b = (int)(row / L);
    int tap = j / 512, c = j % 512;
    int tt = t - 1 + tap; if (tt < 0) tt += L; if (tt >= L) tt -= L;
    dst[i] = (_Float16)X[((size_t)b * L + tt) * DM + c];
}

// WMMA GEMM (generic, 16x16 tile/wave): C = A[N,K]*W[dout_pad,K]^T (+bias)(+res)(act)
__global__ void k_gemm(const _Float16* __restrict__ A, const _Float16* __restrict__ W,
                       const float* __restrict__ bias, const float* __restrict__ res,
                       float* __restrict__ C, int N, int K, int dout, int dout_pad,
                       int act, int ntiles) {
    int tile = blockIdx.x * blockDim.y + threadIdx.y;
    if (tile >= ntiles) return;
    int tilesO = dout_pad >> 4;
    int tm = tile / tilesO;
    int to = tile - tm * tilesO;
    int lane = threadIdx.x;
    int l15 = lane & 15;
    int lhi = lane >> 4;
    const half8* Arow = (const half8*)(A + (size_t)((tm << 4) + l15) * K);
    const half8* Wrow = (const half8*)(W + (size_t)((to << 4) + l15) * K);
    int hiA = lhi << 3;   // 0 or 8
    int hiB = lhi << 4;   // 0 or 16
    v8f acc = {0.f, 0.f, 0.f, 0.f, 0.f, 0.f, 0.f, 0.f};
    for (int k = 0; k < K; k += 32) {
        union { v16h v; half8 h[2]; } a, b;
        a.h[0] = Arow[(k + hiA) >> 3];
        a.h[1] = Arow[(k + hiA + 16) >> 3];
        b.h[0] = Wrow[(k + hiA) >> 3];
        b.h[1] = Wrow[(k + hiA + 16) >> 3];
        acc = __builtin_amdgcn_wmma_f32_16x16x32_f16(false, a.v, false, b.v,
                                                     (short)0, acc, false, false);
    }
    int o = (to << 4) + l15;
    if (o >= dout) return;
    float bv = bias ? bias[o] : 0.f;
    int row0 = (tm << 4) + (lhi << 3);
    for (int r = 0; r < 8; ++r) {
        float vv = acc[r] + bv;
        size_t off = (size_t)(row0 + r) * dout + o;
        if (res) vv += res[off];
        C[off] = act_apply(vv, act);
    }
}

// WMMA GEMM (wide, 16x64 tile/wave, dout_pad % 64 == 0): A fragment shared by 4 WMMAs
__global__ void k_gemm64(const _Float16* __restrict__ A, const _Float16* __restrict__ W,
                         const float* __restrict__ bias, const float* __restrict__ res,
                         float* __restrict__ C, int N, int K, int dout, int dout_pad,
                         int act, int ntiles) {
    int tile = blockIdx.x * blockDim.y + threadIdx.y;
    if (tile >= ntiles) return;
    int tilesO = dout_pad >> 6;
    int tm = tile / tilesO;
    int to = tile - tm * tilesO;
    int lane = threadIdx.x;
    int l15 = lane & 15;
    int lhi = lane >> 4;
    const half8* Arow  = (const half8*)(A + (size_t)((tm << 4) + l15) * K);
    const half8* Wrow0 = (const half8*)(W + (size_t)((to << 6) + l15) * K);
    size_t wstride = ((size_t)16 * K) >> 3;   // 16 W rows, in half8 units
    int hiA = lhi << 3;   // 0 or 8
    int hiB = lhi << 4;   // 0 or 16
    v8f acc[4];
#pragma unroll
    for (int j = 0; j < 4; ++j) acc[j] = (v8f){0.f, 0.f, 0.f, 0.f, 0.f, 0.f, 0.f, 0.f};
    for (int k = 0; k < K; k += 32) {
        union { v16h v; half8 h[2]; } a;
        a.h[0] = Arow[(k + hiA) >> 3];
        a.h[1] = Arow[(k + hiA + 16) >> 3];
#pragma unroll
        for (int j = 0; j < 4; ++j) {
            union { v16h v; half8 h[2]; } b;
            const half8* wr = Wrow0 + (size_t)j * wstride;
            b.h[0] = wr[(k + hiA) >> 3];
            b.h[1] = wr[(k + hiA + 16) >> 3];
            acc[j] = __builtin_amdgcn_wmma_f32_16x16x32_f16(false, a.v, false, b.v,
                                                            (short)0, acc[j], false, false);
        }
    }
    int row0 = (tm << 4) + (lhi << 3);
#pragma unroll
    for (int j = 0; j < 4; ++j) {
        int o = (to << 6) + (j << 4) + l15;
        if (o >= dout) continue;
        float bv = bias ? bias[o] : 0.f;
        for (int r = 0; r < 8; ++r) {
            float vv = acc[j][r] + bv;
            size_t off = (size_t)(row0 + r) * dout + o;
            if (res) vv += res[off];
            C[off] = act_apply(vv, act);
        }
    }
}

// layer norm over last dim (512), one wave per row
__global__ void k_ln(const float* __restrict__ in, float* __restrict__ out,
                     const float* __restrict__ g, const float* __restrict__ bb, int rows) {
    int row = blockIdx.x * blockDim.y + threadIdx.y;
    if (row >= rows) return;
    int lane = threadIdx.x;
    const float* p = in + (size_t)row * DM;
    float s = 0.f;
    for (int i = lane; i < DM; i += 32) s += p[i];
    for (int m = 16; m; m >>= 1) s += __shfl_xor(s, m);
    float mean = s * (1.f / DM);
    float v = 0.f;
    for (int i = lane; i < DM; i += 32) { float d = p[i] - mean; v += d * d; }
    for (int m = 16; m; m >>= 1) v += __shfl_xor(v, m);
    float r = 1.f / sqrtf(v * (1.f / DM) + 1e-5f);
    float* q = out + (size_t)row * DM;
    for (int i = lane; i < DM; i += 32) q[i] = (p[i] - mean) * r * g[i] + bb[i];
}

// JAX threefry sampling: idx = random_bits(split(fold_in(key42,data))[1]) & mask
__global__ void k_rng(int folddata, int n, int mask, int* __restrict__ idx) {
    int i = blockIdx.x * 256 + threadIdx.x;
    if (i >= n) return;
    unsigned a, b;
    tf2x32(0u, 42u, 0u, (unsigned)folddata, a, b);   // layer key = fold_in(key(42), data)
#if JAX_PARTITIONABLE
    unsigned s0, s1; tf2x32(a, b, 0u, 1u, s0, s1);   // k2 = split(key,2)[1]
    unsigned r0, r1; tf2x32(s0, s1, 0u, (unsigned)i, r0, r1);
    unsigned bits = r0 ^ r1;
#else
    unsigned p0, p1, q0, q1;
    tf2x32(a, b, 0u, 2u, p0, p1);
    tf2x32(a, b, 1u, 3u, q0, q1);
    unsigned s0 = p1, s1 = q1;                        // k2 (original split)
    int half = n >> 1;
    int j = (i < half) ? i : i - half;
    unsigned r0, r1; tf2x32(s0, s1, (unsigned)j, (unsigned)(j + half), r0, r1);
    unsigned bits = (i < half) ? r0 : r1;
#endif
    idx[i] = (int)(bits & (unsigned)mask);
}

// M[b,h,q] = max_s(Q.K_sample) - sum_s(Q.K_sample)/L
__global__ void k_qkM(const float* __restrict__ Q, const float* __restrict__ K,
                      const int* __restrict__ idx, float* __restrict__ M, int L, int U) {
    long i = (long)blockIdx.x * 256 + threadIdx.x;
    long total = (long)B * NH * L;
    if (i >= total) return;
    int q = (int)(i % L);
    int h = (int)((i / L) % NH);
    int b = (int)(i / ((long)L * NH));
    const float* qr = Q + ((size_t)b * L + q) * DM + h * DH;
    float mx = -INFINITY, sm = 0.f;
    for (int s = 0; s < U; ++s) {
        const float* kr = K + ((size_t)b * L + idx[q * U + s]) * DM + h * DH;
        float acc = 0.f;
        for (int d = 0; d < DH; ++d) acc += qr[d] * kr[d];
        mx = fmaxf(mx, acc); sm += acc;
    }
    M[i] = mx - sm / (float)L;
}

// top-u selection (stable: lower index wins ties), one block per (b,h)
__global__ void k_topk(const float* __restrict__ M, int* __restrict__ mtop, int L, int U) {
    __shared__ float v[512];
    int bh = blockIdx.x;
    const float* row = M + (size_t)bh * L;
    for (int i = threadIdx.x; i < L; i += blockDim.x) v[i] = row[i];
    __syncthreads();
    if (threadIdx.x == 0) {
        for (int s = 0; s < U; ++s) {
            float best = -INFINITY; int bi = 0;
            for (int i = 0; i < L; ++i) if (v[i] > best) { best = v[i]; bi = i; }
            mtop[bh * U + s] = bi;
            v[bi] = -INFINITY;
        }
    }
}

// scores[bh,ui,k] = scale * Q[mtop].K[k]  (optional causal mask)
__global__ void k_scq(const float* __restrict__ Q, const float* __restrict__ K,
                      const int* __restrict__ mtop, float* __restrict__ sc,
                      int L, int U, int masked, float scale) {
    long i = (long)blockIdx.x * 256 + threadIdx.x;
    long total = (long)B * NH * U * L;
    if (i >= total) return;
    int k  = (int)(i % L);
    int ui = (int)((i / L) % U);
    int bh = (int)(i / ((long)L * U));
    int b = bh / NH, h = bh % NH;
    int qpos = mtop[bh * U + ui];
    const float* qr = Q + ((size_t)b * L + qpos) * DM + h * DH;
    const float* kr = K + ((size_t)b * L + k) * DM + h * DH;
    float acc = 0.f;
    for (int d = 0; d < DH; ++d) acc += qr[d] * kr[d];
    float sv = acc * scale;
    if (masked && k > qpos) sv = -INFINITY;
    sc[(size_t)(bh * U + ui) * L + k] = sv;
}

// row softmax, one wave per row
__global__ void k_sm(float* __restrict__ sc, int rows, int W) {
    int row = blockIdx.x * blockDim.y + threadIdx.y;
    if (row >= rows) return;
    int lane = threadIdx.x;
    float* p = sc + (size_t)row * W;
    float mx = -INFINITY;
    for (int i = lane; i < W; i += 32) mx = fmaxf(mx, p[i]);
    for (int m = 16; m; m >>= 1) mx = fmaxf(mx, __shfl_xor(mx, m));
    float s = 0.f;
    for (int i = lane; i < W; i += 32) { float e = expf(p[i] - mx); p[i] = e; s += e; }
    for (int m = 16; m; m >>= 1) s += __shfl_xor(s, m);
    float inv = 1.f / s;
    for (int i = lane; i < W; i += 32) p[i] *= inv;
}

// upd[bh,ui,d] = attn . V
__global__ void k_upd(const float* __restrict__ sc, const float* __restrict__ V,
                      float* __restrict__ upd, int L, int U) {
    long i = (long)blockIdx.x * 256 + threadIdx.x;
    long total = (long)B * NH * U * DH;
    if (i >= total) return;
    int d  = (int)(i & 63);
    int ui = (int)((i >> 6) % U);
    int bh = (int)(i / ((long)U * DH));
    int b = bh / NH, h = bh % NH;
    const float* ar = sc + (size_t)(bh * U + ui) * L;
    float acc = 0.f;
    for (int k = 0; k < L; ++k) acc += ar[k] * V[((size_t)b * L + k) * DM + h * DH + d];
    upd[i] = acc;
}

__global__ void k_meanv(const float* __restrict__ V, float* __restrict__ mean, int L) {
    int i = blockIdx.x * 256 + threadIdx.x;
    if (i >= B * NH * DH) return;
    int d = i & 63, bh = i >> 6;
    int b = bh / NH, h = bh % NH;
    float acc = 0.f;
    for (int q = 0; q < L; ++q) acc += V[((size_t)b * L + q) * DM + h * DH + d];
    mean[i] = acc / (float)L;
}

__global__ void k_fill(const float* __restrict__ mean, float* __restrict__ ctx, int L, int mix) {
    long i = (long)blockIdx.x * 256 + threadIdx.x;
    long total = (long)B * L * DM;
    if (i >= total) return;
    int c = (int)(i % DM);
    int q = (int)((i / DM) % L);
    int b = (int)(i / ((long)DM * L));
    int h = c >> 6, d = c & 63;
    size_t off = mix ? ((size_t)((b * NH + h) * (long)L + q)) * DH + d : (size_t)i;
    ctx[off] = mean[(b * NH + h) * DH + d];
}

__global__ void k_cumsum(const float* __restrict__ V, float* __restrict__ ctx, int L, int mix) {
    int i = blockIdx.x * 256 + threadIdx.x;
    if (i >= B * NH * DH) return;
    int d = i & 63, bh = i >> 6;
    int b = bh / NH, h = bh % NH;
    float acc = 0.f;
    for (int q = 0; q < L; ++q) {
        acc += V[((size_t)b * L + q) * DM + h * DH + d];
        size_t off = mix ? ((size_t)bh * L + q) * DH + d
                         : ((size_t)(b * L + q)) * DM + h * DH + d;
        ctx[off] = acc;
    }
}

__global__ void k_scatter(const float* __restrict__ upd, const int* __restrict__ mtop,
                          float* __restrict__ ctx, int L, int U, int mix) {
    long i = (long)blockIdx.x * 256 + threadIdx.x;
    long total = (long)B * NH * U * DH;
    if (i >= total) return;
    int d  = (int)(i & 63);
    int ui = (int)((i >> 6) % U);
    int bh = (int)(i / ((long)U * DH));
    int b = bh / NH, h = bh % NH;
    int q = mtop[bh * U + ui];
    size_t off = mix ? ((size_t)bh * L + q) * DH + d
                     : ((size_t)(b * L + q)) * DM + h * DH + d;
    ctx[off] = upd[i];
}

// full attention scores (cross): scale * Q.K
__global__ void k_scf(const float* __restrict__ Q, const float* __restrict__ K,
                      float* __restrict__ sc, int LQ, int LK, float scale) {
    long i = (long)blockIdx.x * 256 + threadIdx.x;
    long total = (long)B * NH * LQ * LK;
    if (i >= total) return;
    int k = (int)(i % LK);
    int q = (int)((i / LK) % LQ);
    int bh = (int)(i / ((long)LK * LQ));
    int b = bh / NH, h = bh % NH;
    const float* qr = Q + ((size_t)b * LQ + q) * DM + h * DH;
    const float* kr = K + ((size_t)b * LK + k) * DM + h * DH;
    float acc = 0.f;
    for (int d = 0; d < DH; ++d) acc += qr[d] * kr[d];
    sc[i] = acc * scale;
}

__global__ void k_ctxf(const float* __restrict__ sc, const float* __restrict__ V,
                       float* __restrict__ ctx, int LQ, int LK) {
    long i = (long)blockIdx.x * 256 + threadIdx.x;
    long total = (long)B * LQ * DM;
    if (i >= total) return;
    int c = (int)(i % DM);
    int q = (int)((i / DM) % LQ);
    int b = (int)(i / ((long)DM * LQ));
    int h = c >> 6, d = c & 63;
    const float* ar = sc + ((size_t)(b * NH + h) * LQ + q) * LK;
    float acc = 0.f;
    for (int k = 0; k < LK; ++k) acc += ar[k] * V[((size_t)b * LK + k) * DM + h * DH + d];
    ctx[i] = acc;
}

__global__ void k_bnelu(const float* __restrict__ in, const float* __restrict__ g,
                        const float* __restrict__ bb, const float* __restrict__ m,
                        const float* __restrict__ v, float* __restrict__ out, long n) {
    long i = (long)blockIdx.x * 256 + threadIdx.x;
    if (i >= n) return;
    int c = (int)(i % DM);
    float y = (in[i] - m[c]) * (1.f / sqrtf(v[c] + 1e-5f)) * g[c] + bb[c];
    out[i] = y > 0.f ? y : expm1f(y);
}

__global__ void k_maxpool(const float* __restrict__ in, float* __restrict__ out, int L) {
    int Lo = L >> 1;
    long i = (long)blockIdx.x * 256 + threadIdx.x;
    long total = (long)B * Lo * DM;
    if (i >= total) return;
    int c = (int)(i % DM);
    long r = i / DM;
    int to = (int)(r % Lo), b = (int)(r / Lo);
    int t = 2 * to;
    float mx = -INFINITY;
    for (int dt = -1; dt <= 1; ++dt) {
        int tt = t + dt;
        if (tt < 0 || tt >= L) continue;
        mx = fmaxf(mx, in[((size_t)b * L + tt) * DM + c]);
    }
    out[((size_t)b * Lo + to) * DM + c] = mx;
}

__global__ void k_slice(const float* __restrict__ proj, float* __restrict__ out) {
    int i = blockIdx.x * 256 + threadIdx.x;
    if (i >= B * 256 * 7) return;
    int c = i % 7, t = (i / 7) % 256, b = i / (7 * 256);
    out[i] = proj[((size_t)(b * 512 + 256 + t)) * 7 + c];
}

// ---------------------------------------------------------------- host side

static inline unsigned gblk(long n) { return (unsigned)((n + 255) / 256); }

static void gemm_launch(hipStream_t st, const _Float16* a16, const _Float16* w16,
                        const float* bias, const float* res, float* out,
                        int N, int Kd, int dout, int dpad, int act) {
    if ((dpad & 63) == 0) {
        int ntiles = (N >> 4) * (dpad >> 6);
        k_gemm64<<<dim3((ntiles + 3) / 4), dim3(32, 4), 0, st>>>(a16, w16, bias, res, out,
                                                                 N, Kd, dout, dpad, act, ntiles);
    } else {
        int ntiles = (N >> 4) * (dpad >> 4);
        k_gemm<<<dim3((ntiles + 3) / 4), dim3(32, 4), 0, st>>>(a16, w16, bias, res, out,
                                                               N, Kd, dout, dpad, act, ntiles);
    }
}

static void lin(hipStream_t st, const float* Ain, int N, int Kd,
                const float* Wp, int dout, const float* bias,
                float* out, const float* res, int act,
                _Float16* a16, _Float16* w16) {
    long na = (long)N * Kd;
    k_f2h<<<gblk(na), 256, 0, st>>>(Ain, a16, na);
    int dpad = (dout + 15) & ~15;
    long nw = (long)dpad * Kd;
    k_w2h<<<gblk(nw), 256, 0, st>>>(Wp, w16, dout, Kd, nw);
    gemm_launch(st, a16, w16, bias, res, out, N, Kd, dout, dpad, act);
}

static void prob_attn(hipStream_t st, const float* Qb, const float* Kb, const float* Vb,
                      float* SC, float* Mb, int* MT, int* IDX, float* UPD, float* MEAN,
                      float* CTX, int L, int U, int folddata, int masked, int mix) {
    k_rng<<<gblk((long)L * U), 256, 0, st>>>(folddata, L * U, L - 1, IDX);
    k_qkM<<<gblk((long)B * NH * L), 256, 0, st>>>(Qb, Kb, IDX, Mb, L, U);
    k_topk<<<dim3(B * NH), dim3(256), 0, st>>>(Mb, MT, L, U);
    k_scq<<<gblk((long)B * NH * U * L), 256, 0, st>>>(Qb, Kb, MT, SC, L, U, masked, 0.125f);
    k_sm<<<dim3((B * NH * U + 7) / 8), dim3(32, 8), 0, st>>>(SC, B * NH * U, L);
    k_upd<<<gblk((long)B * NH * U * DH), 256, 0, st>>>(SC, Vb, UPD, L, U);
    if (masked) {
        k_cumsum<<<gblk((long)B * NH * DH), 256, 0, st>>>(Vb, CTX, L, mix);
    } else {
        k_meanv<<<gblk((long)B * NH * DH), 256, 0, st>>>(Vb, MEAN, L);
        k_fill<<<gblk((long)B * L * DM), 256, 0, st>>>(MEAN, CTX, L, mix);
    }
    k_scatter<<<gblk((long)B * NH * U * DH), 256, 0, st>>>(UPD, MT, CTX, L, U, mix);
}

extern "C" void kernel_launch(void* const* d_in, const int* in_sizes, int n_in,
                              void* d_out, int out_size, void* d_ws, size_t ws_size,
                              hipStream_t stream) {
    (void)in_sizes; (void)out_size; (void)ws_size;
    if (n_in < 144) return;
    auto F = [&](int i) -> const float* { return (const float*)d_in[i]; };
    auto I = [&](int i) -> const int*   { return (const int*)d_in[i]; };

    char* ws = (char*)d_ws;
    const size_t MB = 1024 * 1024;
    _Float16* A16 = (_Float16*)(ws + 0);        // 16 MB
    _Float16* W16 = (_Float16*)(ws + 16 * MB);  // 4 MB
    float* X    = (float*)(ws + 20 * MB);
    float* Y    = (float*)(ws + 28 * MB);
    float* T    = (float*)(ws + 36 * MB);
    float* Qb   = (float*)(ws + 44 * MB);
    float* Kb   = (float*)(ws + 52 * MB);
    float* Vb   = (float*)(ws + 60 * MB);
    float* CTX  = (float*)(ws + 68 * MB);
    float* FFB  = (float*)(ws + 76 * MB);       // 32 MB
    float* ENC  = (float*)(ws + 108 * MB);
    float* SC   = (float*)(ws + 116 * MB);      // 16 MB
    float* Mb   = (float*)(ws + 132 * MB);
    int*   MT   = (int*)  (ws + 133 * MB);
    int*   IDX  = (int*)  (ws + 134 * MB);
    float* UPD  = (float*)(ws + 135 * MB);
    float* MEAN = (float*)(ws + 136 * MB);
    float* ADPT = (float*)(ws + 137 * MB);      // 2 MB
    float* PROJ = (float*)(ws + 139 * MB);

    // param leaf indices (params tree, sorted dict keys; lists in order)
    // 0: dec_emb.conv_W ; 1+30l: dec layer l ; 61/62: dec_norm b/g ;
    // 63+6j: enc_convs[j] {W,b,bn_b,bn_g,bn_m,bn_v} ; 75: enc_emb.conv_W ;
    // 76+20l: enc layer l ; 136/137: enc_norm b/g ; 138/139: proj W/b ;
    // 140 x_enc, 141 x_dec, 142 x_mark_enc, 143 x_mark_dec
    const float* x_enc = F(140);
    const float* x_dec = F(141);
    const int* mk_enc  = I(142);
    const int* mk_dec  = I(143);

    // -------------------- encoder --------------------
    int Ls[3] = {512, 256, 128};
    int Us[3] = {35, 30, 25};
    k_embed<<<gblk((long)B * 512 * DM), 256, 0, stream>>>(x_enc, mk_enc, F(75), X, 512);
    for (int l = 0; l < 3; ++l) {
        int L = Ls[l], N = B * L, U = Us[l];
        int e = 76 + 20 * l;
        // prob self-attention (mask=false, mix=false)
        lin(stream, X, N, DM, F(e + 8),  DM, F(e + 9),  Qb, nullptr, 0, A16, W16);
        lin(stream, X, N, DM, F(e + 4),  DM, F(e + 5),  Kb, nullptr, 0, A16, W16);
        lin(stream, X, N, DM, F(e + 10), DM, F(e + 11), Vb, nullptr, 0, A16, W16);
        prob_attn(stream, Qb, Kb, Vb, SC, Mb, MT, IDX, UPD, MEAN, CTX, L, U, l, 0, 0);
        lin(stream, CTX, N, DM, F(e + 6), DM, F(e + 7), Y, X, 0, A16, W16);          // o + residual
        k_ln<<<dim3((N + 7) / 8), dim3(32, 8), 0, stream>>>(Y, X, F(e + 17), F(e + 16), N); // norm1
        // FFN
        lin(stream, X, N, DM, F(e + 12), DFF, F(e + 13), FFB, nullptr, 2, A16, W16); // ff1 + gelu
        lin(stream, FFB, N, DFF, F(e + 14), DM, F(e + 15), Y, X, 0, A16, W16);       // ff2 + residual
        k_ln<<<dim3((N + 7) / 8), dim3(32, 8), 0, stream>>>(Y, T, F(e + 19), F(e + 18), N); // norm2
        // adapter
        lin(stream, T, N, DM, F(e + 0), 64, F(e + 1), ADPT, nullptr, 1, A16, W16);   // down + relu
        lin(stream, ADPT, N, 64, F(e + 2), DM, F(e + 3), X, T, 0, A16, W16);         // up + residual
        if (l < 2) {
            int cb = 63 + 6 * l;
            k_build3<<<gblk((long)N * 1536), 256, 0, stream>>>(X, A16, L);
            k_packW<<<gblk(512 * 1536), 256, 0, stream>>>(F(cb + 0), W16);
            gemm_launch(stream, A16, W16, F(cb + 1), nullptr, Y, N, 1536, DM, DM, 0);
            k_bnelu<<<gblk((long)N * DM), 256, 0, stream>>>(Y, F(cb + 3), F(cb + 2),
                                                            F(cb + 4), F(cb + 5), Y, (long)N * DM);
            k_maxpool<<<gblk((long)B * (L / 2) * DM), 256, 0, stream>>>(Y, X, L);
        }
    }
    k_ln<<<dim3((B * 128 + 7) / 8), dim3(32, 8), 0, stream>>>(X, ENC, F(137), F(136), B * 128);

    // -------------------- decoder --------------------
    const int LD = 512, ND = B * LD, LE = 128, NE = B * LE;
    k_embed<<<gblk((long)ND * DM), 256, 0, stream>>>(x_dec, mk_dec, F(0), X, LD);
    for (int l = 0; l < 2; ++l) {
        int d = 1 + 30 * l;
        // masked prob self-attention (mix=true)
        lin(stream, X, ND, DM, F(d + 26), DM, F(d + 27), Qb, nullptr, 0, A16, W16);
        lin(stream, X, ND, DM, F(d + 22), DM, F(d + 23), Kb, nullptr, 0, A16, W16);
        lin(stream, X, ND, DM, F(d + 28), DM, F(d + 29), Vb, nullptr, 0, A16, W16);
        prob_attn(stream, Qb, Kb, Vb, SC, Mb, MT, IDX, UPD, MEAN, CTX, LD, 35, 100 + l, 1, 1);
        lin(stream, CTX, ND, DM, F(d + 24), DM, F(d + 25), Y, X, 0, A16, W16);
        k_ln<<<dim3((ND + 7) / 8), dim3(32, 8), 0, stream>>>(Y, X, F(d + 17), F(d + 16), ND);
        // full cross-attention
        lin(stream, X,   ND, DM, F(d + 8),  DM, F(d + 9),  Qb, nullptr, 0, A16, W16);
        lin(stream, ENC, NE, DM, F(d + 4),  DM, F(d + 5),  Kb, nullptr, 0, A16, W16);
        lin(stream, ENC, NE, DM, F(d + 10), DM, F(d + 11), Vb, nullptr, 0, A16, W16);
        k_scf<<<gblk((long)B * NH * LD * LE), 256, 0, stream>>>(Qb, Kb, SC, LD, LE, 0.125f);
        k_sm<<<dim3((B * NH * LD + 7) / 8), dim3(32, 8), 0, stream>>>(SC, B * NH * LD, LE);
        k_ctxf<<<gblk((long)B * LD * DM), 256, 0, stream>>>(SC, Vb, CTX, LD, LE);
        lin(stream, CTX, ND, DM, F(d + 6), DM, F(d + 7), Y, X, 0, A16, W16);
        k_ln<<<dim3((ND + 7) / 8), dim3(32, 8), 0, stream>>>(Y, X, F(d + 19), F(d + 18), ND);
        // FFN
        lin(stream, X, ND, DM, F(d + 12), DFF, F(d + 13), FFB, nullptr, 2, A16, W16);
        lin(stream, FFB, ND, DFF, F(d + 14), DM, F(d + 15), Y, X, 0, A16, W16);
        k_ln<<<dim3((ND + 7) / 8), dim3(32, 8), 0, stream>>>(Y, T, F(d + 21), F(d + 20), ND);
        // adapter
        lin(stream, T, ND, DM, F(d + 0), 64, F(d + 1), ADPT, nullptr, 1, A16, W16);
        lin(stream, ADPT, ND, 64, F(d + 2), DM, F(d + 3), X, T, 0, A16, W16);
    }
    k_ln<<<dim3((ND + 7) / 8), dim3(32, 8), 0, stream>>>(X, Y, F(62), F(61), ND);
    lin(stream, Y, ND, DM, F(138), 7, F(139), PROJ, nullptr, 0, A16, W16);
    k_slice<<<gblk(B * 256 * 7), 256, 0, stream>>>(PROJ, (float*)d_out);
}
